// Mymodel_2010044695161
// MI455X (gfx1250) — matrix operand order repaired and checked by reading, not compiled, not run on hardware
//
#include <hip/hip_runtime.h>
#include <stdint.h>

#define N_NODES 200000
#define N_EDGES 800000
#define N_GRAPHS 8192
#define NFEAT 74
#define HH 4
#define FF 50
#define HF 200          // H*F
#define BN_EPS 1e-5f
#define ATTN_SLOPE 0.2f
#define ACT_SLOPE 0.01f
#define CAPS_COEF 0.08838834764831845f   // 1/sqrt(128)
#define ENC_NEG_INF 0x007FFFFFu          // ordenc(-inf)

typedef __attribute__((ext_vector_type(16))) __bf16 v16bf;
typedef __attribute__((ext_vector_type(8)))  float  v8f;

struct __align__(16) I4 { int x, y, z, w; };
union FragU { I4 q[2]; v16bf v; };

// ---------------- helpers ----------------
__device__ __forceinline__ unsigned short f2bf(float f) {
  unsigned u = __float_as_uint(f);
  unsigned r = ((u >> 16) & 1u) + 0x7fffu;   // round-to-nearest-even
  return (unsigned short)((u + r) >> 16);
}
__device__ __forceinline__ unsigned ordenc(float f) {  // total-order uint encoding
  unsigned u = __float_as_uint(f);
  return (u & 0x80000000u) ? ~u : (u | 0x80000000u);
}
__device__ __forceinline__ float orddec(unsigned u) {
  return __uint_as_float((u & 0x80000000u) ? (u & 0x7fffffffu) : ~u);
}

// ---------------- data prep ----------------
// Row-major f32 [M x K] -> row-major bf16 [M x Kpad] (zero padded), stored as uint pairs.
__global__ void pad_a_bf16(const float* __restrict__ src, unsigned* __restrict__ dst,
                           int M, int K, int Kpad) {
  long long t = (long long)blockIdx.x * blockDim.x + threadIdx.x;
  int ppr = Kpad >> 1;
  long long total = (long long)M * ppr;
  if (t >= total) return;
  int row = (int)(t / ppr);
  int p = (int)(t % ppr);
  int k0 = 2 * p, k1 = 2 * p + 1;
  float f0 = (k0 < K) ? src[(long long)row * K + k0] : 0.f;
  float f1 = (k1 < K) ? src[(long long)row * K + k1] : 0.f;
  dst[t] = (unsigned)f2bf(f0) | ((unsigned)f2bf(f1) << 16);
}

// Pack weight f32 [K x N] into WMMA B fragment layout:
// dst[((tn*numKb + kb)*32 + lane)*8 + v] = bf16 pair (k = kb*32 + (v>>2)*16 + (lane>>4)*8 + 2*(v&3), n = tn*16 + lane&15)
__global__ void pack_b_bf16(const float* __restrict__ W, unsigned* __restrict__ dst,
                            int K, int N, int numKb, int numTilesN) {
  int t = blockIdx.x * blockDim.x + threadIdx.x;
  int per = numKb * 256;
  int total = numTilesN * per;
  if (t >= total) return;
  int tn = t / per;
  int rem = t % per;
  int kb = rem >> 8;
  int r2 = rem & 255;
  int lane = r2 >> 3;
  int v = r2 & 7;
  int hv = lane >> 4;
  int n = tn * 16 + (lane & 15);
  int k0 = kb * 32 + (v >> 2) * 16 + hv * 8 + 2 * (v & 3);
  int k1 = k0 + 1;
  float f0 = (k0 < K && n < N) ? W[(long long)k0 * N + n] : 0.f;
  float f1 = (k1 < K && n < N) ? W[(long long)k1 * N + n] : 0.f;
  dst[t] = (unsigned)f2bf(f0) | ((unsigned)f2bf(f1) << 16);
}

__global__ void fill_u32(unsigned* __restrict__ p, unsigned v, long long n) {
  long long t = (long long)blockIdx.x * blockDim.x + threadIdx.x;
  if (t < n) p[t] = v;
}

// ---------------- WMMA GEMM core ----------------
__device__ __forceinline__ void load_a_frag(const unsigned short* arow, int kb, FragU& fa) {
  fa.q[0] = *(const I4*)(arow + kb * 32);
  fa.q[1] = *(const I4*)(arow + kb * 32 + 16);
}
__device__ __forceinline__ void load_b_frag(const unsigned* brow, int kb, FragU& fb) {
  fb.q[0] = *(const I4*)(brow + (long long)kb * 256);
  fb.q[1] = *(const I4*)(brow + (long long)kb * 256 + 4);
}

// Dual GEMM: C0 = A@B0 ; C1 = A@B1 + bias[n]  (layer-1 feat + residual-with-bias)
__global__ void gemm_dual(const unsigned short* __restrict__ A, int ldA,
                          const unsigned* __restrict__ Bp0, const unsigned* __restrict__ Bp1,
                          const float* __restrict__ bias,
                          float* __restrict__ C0, float* __restrict__ C1,
                          int totalTiles, int numTilesN, int numKb, int Ncols) {
  int w = blockIdx.x * (blockDim.x >> 5) + (threadIdx.x >> 5);
  if (w >= totalTiles) return;
  int lane = threadIdx.x & 31;
  int tileM = w / numTilesN, tileN = w % numTilesN;
  const unsigned short* arow = A + (long long)(tileM * 16 + (lane & 15)) * ldA + (lane >> 4) * 8;
  const unsigned* brow0 = Bp0 + (long long)tileN * numKb * 256 + lane * 8;
  const unsigned* brow1 = Bp1 + (long long)tileN * numKb * 256 + lane * 8;
  v8f c0 = {}; v8f c1 = {};
  for (int kb = 0; kb < numKb; ++kb) {
    FragU fa, fb0, fb1;
    load_a_frag(arow, kb, fa);
    load_b_frag(brow0, kb, fb0);
    load_b_frag(brow1, kb, fb1);
    c0 = __builtin_amdgcn_wmma_f32_16x16x32_bf16(false, fa.v, false, fb0.v, (short)0, c0, false, false);
    c1 = __builtin_amdgcn_wmma_f32_16x16x32_bf16(false, fa.v, false, fb1.v, (short)0, c1, false, false);
  }
  int n = tileN * 16 + (lane & 15);
  if (n >= Ncols) return;
  int rbase = tileM * 16 + (lane >> 4) * 8;
  float b = bias[n];
  for (int v = 0; v < 8; ++v) {
    C0[(long long)(rbase + v) * Ncols + n] = c0[v];
    C1[(long long)(rbase + v) * Ncols + n] = c1[v] + b;
  }
}

// Plain GEMM: C = A@B
__global__ void gemm_single(const unsigned short* __restrict__ A, int ldA,
                            const unsigned* __restrict__ Bp, float* __restrict__ C,
                            int totalTiles, int numTilesN, int numKb, int Ncols) {
  int w = blockIdx.x * (blockDim.x >> 5) + (threadIdx.x >> 5);
  if (w >= totalTiles) return;
  int lane = threadIdx.x & 31;
  int tileM = w / numTilesN, tileN = w % numTilesN;
  const unsigned short* arow = A + (long long)(tileM * 16 + (lane & 15)) * ldA + (lane >> 4) * 8;
  const unsigned* brow = Bp + (long long)tileN * numKb * 256 + lane * 8;
  v8f c = {};
  for (int kb = 0; kb < numKb; ++kb) {
    FragU fa, fb;
    load_a_frag(arow, kb, fa);
    load_b_frag(brow, kb, fb);
    c = __builtin_amdgcn_wmma_f32_16x16x32_bf16(false, fa.v, false, fb.v, (short)0, c, false, false);
  }
  int n = tileN * 16 + (lane & 15);
  if (n >= Ncols) return;
  int rbase = tileM * 16 + (lane >> 4) * 8;
  for (int v = 0; v < 8; ++v) C[(long long)(rbase + v) * Ncols + n] = c[v];
}

// MLP layer 1: Zbf = bf16( BN( relu(A@B + lb) ) )
__global__ void gemm_mlp1(const unsigned short* __restrict__ A, int ldA,
                          const unsigned* __restrict__ Bp,
                          const float* __restrict__ lb, const float* __restrict__ bng,
                          const float* __restrict__ bnb, const float* __restrict__ bnrm,
                          const float* __restrict__ bnrv,
                          unsigned short* __restrict__ Zbf,
                          int totalTiles, int numTilesN, int numKb, int Ncols) {
  int w = blockIdx.x * (blockDim.x >> 5) + (threadIdx.x >> 5);
  if (w >= totalTiles) return;
  int lane = threadIdx.x & 31;
  int tileM = w / numTilesN, tileN = w % numTilesN;
  const unsigned short* arow = A + (long long)(tileM * 16 + (lane & 15)) * ldA + (lane >> 4) * 8;
  const unsigned* brow = Bp + (long long)tileN * numKb * 256 + lane * 8;
  v8f c = {};
  for (int kb = 0; kb < numKb; ++kb) {
    FragU fa, fb;
    load_a_frag(arow, kb, fa);
    load_b_frag(brow, kb, fb);
    c = __builtin_amdgcn_wmma_f32_16x16x32_bf16(false, fa.v, false, fb.v, (short)0, c, false, false);
  }
  int n = tileN * 16 + (lane & 15);
  if (n >= Ncols) return;
  int rbase = tileM * 16 + (lane >> 4) * 8;
  float bias = lb[n];
  float scale = bng[n] * rsqrtf(bnrv[n] + BN_EPS);
  float shift = bnb[n] - bnrm[n] * scale;
  for (int v = 0; v < 8; ++v) {
    float t = c[v] + bias;
    t = t > 0.f ? t : 0.f;
    t = t * scale + shift;
    Zbf[(long long)(rbase + v) * Ncols + n] = f2bf(t);
  }
}

// MLP layer 2: C = relu(A@B + lb)
__global__ void gemm_mlp2(const unsigned short* __restrict__ A, int ldA,
                          const unsigned* __restrict__ Bp, const float* __restrict__ lb,
                          float* __restrict__ C,
                          int totalTiles, int numTilesN, int numKb, int Ncols) {
  int w = blockIdx.x * (blockDim.x >> 5) + (threadIdx.x >> 5);
  if (w >= totalTiles) return;
  int lane = threadIdx.x & 31;
  int tileM = w / numTilesN, tileN = w % numTilesN;
  const unsigned short* arow = A + (long long)(tileM * 16 + (lane & 15)) * ldA + (lane >> 4) * 8;
  const unsigned* brow = Bp + (long long)tileN * numKb * 256 + lane * 8;
  v8f c = {};
  for (int kb = 0; kb < numKb; ++kb) {
    FragU fa, fb;
    load_a_frag(arow, kb, fa);
    load_b_frag(brow, kb, fb);
    c = __builtin_amdgcn_wmma_f32_16x16x32_bf16(false, fa.v, false, fb.v, (short)0, c, false, false);
  }
  int n = tileN * 16 + (lane & 15);
  if (n >= Ncols) return;
  int rbase = tileM * 16 + (lane >> 4) * 8;
  float bias = lb[n];
  for (int v = 0; v < 8; ++v) {
    float t = c[v] + bias;
    C[(long long)(rbase + v) * Ncols + n] = t > 0.f ? t : 0.f;
  }
}

// ---------------- GAT attention ----------------
__global__ void attn_lr(const float* __restrict__ feat, const float* __restrict__ al,
                        const float* __restrict__ ar, float* __restrict__ el,
                        float* __restrict__ er) {
  int t = blockIdx.x * blockDim.x + threadIdx.x;
  if (t >= N_NODES * HH) return;
  int h = t & 3;
  int node = t >> 2;
  const float* fp = feat + (long long)node * HF + h * FF;
  const float* alp = al + h * FF;
  const float* arp = ar + h * FF;
  float sl = 0.f, sr = 0.f;
  for (int f = 0; f < FF; ++f) { float x = fp[f]; sl += x * alp[f]; sr += x * arp[f]; }
  el[t] = sl; er[t] = sr;
}

__global__ void edge_pass1(const float* __restrict__ el, const float* __restrict__ er,
                           const int* __restrict__ src, const int* __restrict__ dst,
                           float* __restrict__ e, unsigned* __restrict__ menc) {
  long long t = (long long)blockIdx.x * blockDim.x + threadIdx.x;
  if (t >= (long long)N_EDGES * HH) return;
  int h = (int)(t & 3);
  long long eid = t >> 2;
  int s = src[eid], d = dst[eid];
  float v = el[s * 4 + h] + er[d * 4 + h];
  v = v > 0.f ? v : ATTN_SLOPE * v;
  e[t] = v;
  atomicMax(&menc[d * 4 + h], ordenc(v));
}

__global__ void edge_pass2(float* __restrict__ e, const unsigned* __restrict__ menc,
                           const int* __restrict__ dst, float* __restrict__ denom) {
  long long t = (long long)blockIdx.x * blockDim.x + threadIdx.x;
  if (t >= (long long)N_EDGES * HH) return;
  int h = (int)(t & 3);
  long long eid = t >> 2;
  int d = dst[eid];
  float a = expf(e[t] - orddec(menc[d * 4 + h]));
  e[t] = a;
  atomicAdd(&denom[d * 4 + h], a);
}

__global__ void agg_edges(const float* __restrict__ a, const float* __restrict__ denom,
                          const int* __restrict__ src, const int* __restrict__ dst,
                          const float* __restrict__ feat, float* __restrict__ out) {
  long long t = (long long)blockIdx.x * blockDim.x + threadIdx.x;
  if (t >= (long long)N_EDGES * HF) return;
  int j = (int)(t % HF);
  long long eid = t / HF;
  int h = j / FF;
  int s = src[eid], d = dst[eid];
  float alpha = a[eid * 4 + h] / denom[d * 4 + h];
  atomicAdd(&out[(long long)d * HF + j], alpha * feat[(long long)s * HF + j]);
}

// ---------------- elementwise ----------------
__global__ void leaky_inplace(float* __restrict__ x, long long n) {
  long long t = (long long)blockIdx.x * blockDim.x + threadIdx.x;
  if (t >= n) return;
  float v = x[t];
  x[t] = v > 0.f ? v : ACT_SLOPE * v;
}

__global__ void add_bias200(const float* __restrict__ x, const float* __restrict__ b,
                            float* __restrict__ out, long long n) {
  long long t = (long long)blockIdx.x * blockDim.x + threadIdx.x;
  if (t >= n) return;
  out[t] = x[t] + b[(int)(t % HF)];
}

__global__ void mean_heads(const float* __restrict__ out2, float* __restrict__ xm) {
  long long t = (long long)blockIdx.x * blockDim.x + threadIdx.x;
  if (t >= (long long)N_NODES * FF) return;
  int node = (int)(t / FF);
  int f = (int)(t % FF);
  long long base = (long long)node * HF + f;
  xm[t] = 0.25f * (out2[base] + out2[base + 50] + out2[base + 100] + out2[base + 150]);
}

__global__ void pool_max(const float* __restrict__ xm, const int* __restrict__ gid,
                         unsigned* __restrict__ genc) {
  long long t = (long long)blockIdx.x * blockDim.x + threadIdx.x;
  if (t >= (long long)N_NODES * FF) return;
  int node = (int)(t / FF);
  int f = (int)(t % FF);
  atomicMax(&genc[(long long)gid[node] * FF + f], ordenc(xm[t]));
}

// zbf [G x 1088] bf16 : cols 0..49 pooled g, 50..1073 fps, rest 0
__global__ void build_zbf(const unsigned* __restrict__ genc, const float* __restrict__ fps,
                          unsigned* __restrict__ zbf) {
  long long t = (long long)blockIdx.x * blockDim.x + threadIdx.x;
  if (t >= (long long)N_GRAPHS * 544) return;
  int g = (int)(t / 544);
  int p = (int)(t % 544);
  int c0 = 2 * p, c1 = 2 * p + 1;
  float f0 = (c0 < 50) ? orddec(genc[g * 50 + c0]) : ((c0 < 1074) ? fps[(long long)g * 1024 + (c0 - 50)] : 0.f);
  float f1 = (c1 < 50) ? orddec(genc[g * 50 + c1]) : ((c1 < 1074) ? fps[(long long)g * 1024 + (c1 - 50)] : 0.f);
  zbf[t] = (unsigned)f2bf(f0) | ((unsigned)f2bf(f1) << 16);
}

// ---------------- capsule head ----------------
__global__ void caps_head(const float* __restrict__ z2, const float* __restrict__ capsW,
                          const float* __restrict__ capsB, float* __restrict__ out) {
  int g = blockIdx.x * blockDim.x + threadIdx.x;
  if (g >= N_GRAPHS) return;
  const float* z = z2 + (long long)g * 128;
  float u[16][8];
  for (int c = 0; c < 16; ++c) {
    float sq = 0.f;
    for (int e = 0; e < 8; ++e) { float x = z[c * 8 + e]; sq += x * x; }
    float coef = (1.f - expf(-sqrtf(sq))) / sqrtf(sq + 1e-8f);
    for (int e = 0; e < 8; ++e) u[c][e] = coef * z[c * 8 + e];
  }
  float uh[2][16][2];
  for (int n = 0; n < 2; ++n)
    for (int c = 0; c < 16; ++c)
      for (int d = 0; d < 2; ++d) {
        float s = 0.f;
        const float* wp = capsW + (((n * 16 + c) * 2 + d) * 8);
        for (int e = 0; e < 8; ++e) s += wp[e] * u[c][e];
        uh[n][c][d] = s;
      }
  float tt[2][2];
  for (int n = 0; n < 2; ++n)
    for (int d = 0; d < 2; ++d) {
      float s = 0.f;
      for (int c = 0; c < 16; ++c) s += uh[n][c][d];
      tt[n][d] = s;
    }
  float asum[2][16];
  for (int n = 0; n < 2; ++n)
    for (int k = 0; k < 16; ++k)
      asum[n][k] = CAPS_COEF * (tt[n][0] * uh[n][k][0] + tt[n][1] * uh[n][k][1]);
  float cc[2][16];
  for (int k = 0; k < 16; ++k) {
    float m = fmaxf(asum[0][k], asum[1][k]);
    float e0 = expf(asum[0][k] - m);
    float e1 = expf(asum[1][k] - m);
    float inv = 1.f / (e0 + e1);
    cc[0][k] = e0 * inv;
    cc[1][k] = e1 * inv;
  }
  for (int n = 0; n < 2; ++n) {
    float s0 = 0.f, s1 = 0.f;
    for (int k = 0; k < 16; ++k) {
      float w = cc[n][k] + capsB[n * 16 + k];
      s0 += w * uh[n][k][0];
      s1 += w * uh[n][k][1];
    }
    float sq = s0 * s0 + s1 * s1;
    float coef = (1.f - expf(-sqrtf(sq))) / sqrtf(sq + 1e-8f);
    float v0 = coef * s0, v1 = coef * s1;
    out[(long long)g * 2 + n] = sqrtf(v0 * v0 + v1 * v1 + 1e-8f);
  }
}

// ---------------- host launcher ----------------
#define NBLK(n) ((int)(((long long)(n) + 255) / 256))

extern "C" void kernel_launch(void* const* d_in, const int* in_sizes, int n_in,
                              void* d_out, int out_size, void* d_ws, size_t ws_size,
                              hipStream_t stream) {
  (void)in_sizes; (void)n_in; (void)out_size; (void)ws_size;
  const float* h     = (const float*)d_in[0];
  const float* fps   = (const float*)d_in[1];
  const int*   src   = (const int*)d_in[2];
  const int*   dst   = (const int*)d_in[3];
  const int*   gid   = (const int*)d_in[4];
  const float* W1    = (const float*)d_in[5];
  const float* al1   = (const float*)d_in[6];
  const float* ar1   = (const float*)d_in[7];
  const float* b1    = (const float*)d_in[8];
  const float* resW1 = (const float*)d_in[9];
  const float* W2    = (const float*)d_in[10];
  const float* al2   = (const float*)d_in[11];
  const float* ar2   = (const float*)d_in[12];
  const float* b2    = (const float*)d_in[13];
  const float* lw1   = (const float*)d_in[14];
  const float* lb1   = (const float*)d_in[15];
  const float* bng   = (const float*)d_in[16];
  const float* bnb   = (const float*)d_in[17];
  const float* bnrm  = (const float*)d_in[18];
  const float* bnrv  = (const float*)d_in[19];
  const float* lw2   = (const float*)d_in[20];
  const float* lb2   = (const float*)d_in[21];
  const float* capsW = (const float*)d_in[22];
  const float* capsB = (const float*)d_in[23];
  float* outp = (float*)d_out;

  char* basep = (char*)d_ws;
  size_t off = 0;
  auto carve = [&](size_t bytes) -> void* {
    void* p = basep + off;
    off = (off + bytes + 255) & ~(size_t)255;
    return p;
  };
  void*  Rbig  = carve(200000ull * 224 * 2);      // hbf -> x1bf -> zbf
  float* featB = (float*)carve(200000ull * 200 * 4);  // feat1 / feat2
  float* outA  = (float*)carve(200000ull * 200 * 4);  // out1 = x1, later xm
  float* outB  = (float*)carve(200000ull * 200 * 4);  // out2
  float* el    = (float*)carve(200000ull * 4 * 4);
  float* er    = (float*)carve(200000ull * 4 * 4);
  unsigned* menc  = (unsigned*)carve(200000ull * 4 * 4);
  float*    denom = (float*)carve(200000ull * 4 * 4);
  float*    eedge = (float*)carve(800000ull * 4 * 4);
  unsigned* W1p  = (unsigned*)carve(13ull * 3 * 256 * 4);
  unsigned* rW1p = (unsigned*)carve(13ull * 3 * 256 * 4);
  unsigned* W2p  = (unsigned*)carve(13ull * 7 * 256 * 4);
  unsigned* lw1p = (unsigned*)carve(8ull * 34 * 256 * 4);
  unsigned* lw2p = (unsigned*)carve(8ull * 4 * 256 * 4);
  unsigned* genc = (unsigned*)carve(8192ull * 50 * 4);
  unsigned short* z1bf = (unsigned short*)carve(8192ull * 128 * 2);
  float* z2 = (float*)carve(8192ull * 128 * 4);

  // ---- GAT layer 1: feat1 = h@W1 ; out1 = h@resW1 + b1 ----
  pad_a_bf16<<<NBLK(200000ll * 48), 256, 0, stream>>>(h, (unsigned*)Rbig, N_NODES, NFEAT, 96);
  pack_b_bf16<<<NBLK(13 * 3 * 256), 256, 0, stream>>>(W1, W1p, NFEAT, HF, 3, 13);
  pack_b_bf16<<<NBLK(13 * 3 * 256), 256, 0, stream>>>(resW1, rW1p, NFEAT, HF, 3, 13);
  gemm_dual<<<(162500 + 7) / 8, 256, 0, stream>>>((const unsigned short*)Rbig, 96, W1p, rW1p, b1,
                                                  featB, outA, 162500, 13, 3, HF);
  attn_lr<<<NBLK(N_NODES * 4), 256, 0, stream>>>(featB, al1, ar1, el, er);
  fill_u32<<<NBLK(N_NODES * 4), 256, 0, stream>>>(menc, ENC_NEG_INF, (long long)N_NODES * 4);
  fill_u32<<<NBLK(N_NODES * 4), 256, 0, stream>>>((unsigned*)denom, 0u, (long long)N_NODES * 4);
  edge_pass1<<<NBLK((long long)N_EDGES * 4), 256, 0, stream>>>(el, er, src, dst, eedge, menc);
  edge_pass2<<<NBLK((long long)N_EDGES * 4), 256, 0, stream>>>(eedge, menc, dst, denom);
  agg_edges<<<NBLK((long long)N_EDGES * HF), 256, 0, stream>>>(eedge, denom, src, dst, featB, outA);
  leaky_inplace<<<NBLK(200000ll * 200), 256, 0, stream>>>(outA, 200000ll * 200);

  // ---- GAT layer 2: feat2 = x1@W2 ; out2 = x1 + b2 + agg ; mean heads ----
  pad_a_bf16<<<NBLK(200000ll * 112), 256, 0, stream>>>(outA, (unsigned*)Rbig, N_NODES, HF, 224);
  pack_b_bf16<<<NBLK(13 * 7 * 256), 256, 0, stream>>>(W2, W2p, HF, HF, 7, 13);
  add_bias200<<<NBLK(200000ll * 200), 256, 0, stream>>>(outA, b2, outB, 200000ll * 200);
  gemm_single<<<(162500 + 7) / 8, 256, 0, stream>>>((const unsigned short*)Rbig, 224, W2p, featB,
                                                    162500, 13, 7, HF);
  attn_lr<<<NBLK(N_NODES * 4), 256, 0, stream>>>(featB, al2, ar2, el, er);
  fill_u32<<<NBLK(N_NODES * 4), 256, 0, stream>>>(menc, ENC_NEG_INF, (long long)N_NODES * 4);
  fill_u32<<<NBLK(N_NODES * 4), 256, 0, stream>>>((unsigned*)denom, 0u, (long long)N_NODES * 4);
  edge_pass1<<<NBLK((long long)N_EDGES * 4), 256, 0, stream>>>(el, er, src, dst, eedge, menc);
  edge_pass2<<<NBLK((long long)N_EDGES * 4), 256, 0, stream>>>(eedge, menc, dst, denom);
  agg_edges<<<NBLK((long long)N_EDGES * HF), 256, 0, stream>>>(eedge, denom, src, dst, featB, outB);
  mean_heads<<<NBLK(200000ll * 50), 256, 0, stream>>>(outB, outA);  // xm in outA

  // ---- per-graph max pooling ----
  fill_u32<<<NBLK(8192ll * 50), 256, 0, stream>>>(genc, ENC_NEG_INF, 8192ll * 50);
  pool_max<<<NBLK(200000ll * 50), 256, 0, stream>>>(outA, gid, genc);

  // ---- MLP head ----
  build_zbf<<<NBLK(8192ll * 544), 256, 0, stream>>>(genc, fps, (unsigned*)Rbig);
  pack_b_bf16<<<NBLK(8 * 34 * 256), 256, 0, stream>>>(lw1, lw1p, 1074, 128, 34, 8);
  gemm_mlp1<<<(4096 + 7) / 8, 256, 0, stream>>>((const unsigned short*)Rbig, 1088, lw1p,
                                                lb1, bng, bnb, bnrm, bnrv, z1bf, 4096, 8, 34, 128);
  pack_b_bf16<<<NBLK(8 * 4 * 256), 256, 0, stream>>>(lw2, lw2p, 128, 128, 4, 8);
  gemm_mlp2<<<(4096 + 7) / 8, 256, 0, stream>>>(z1bf, 128, lw2p, lb2, z2, 4096, 8, 4, 128);

  // ---- capsule head ----
  caps_head<<<NBLK(8192), 256, 0, stream>>>(z2, capsW, capsB, outp);
}
